// AttnBlock_45586782880022
// MI455X (gfx1250) — matrix operand order repaired and checked by reading, not compiled, not run on hardware
//
#include <hip/hip_runtime.h>
#include <stdint.h>

typedef __attribute__((ext_vector_type(16))) __bf16       v16bf;
typedef __attribute__((ext_vector_type(8)))  float        v8f;
typedef __attribute__((ext_vector_type(4)))  unsigned int u32x4;
typedef __attribute__((ext_vector_type(2)))  unsigned int u32x2;
typedef unsigned short ushort_t;

#define C_      512
#define HW_     4096
#define BATCH_  8
#define GROUPS_ 32
#define MTOT    (BATCH_ * HW_)   // 32768

// ---- CDNA5 async global->LDS staging (guarded; falls back to VGPR copies) ----
#if defined(__HIP_DEVICE_COMPILE__) &&                                   \
    __has_builtin(__builtin_amdgcn_global_load_async_to_lds_b128) &&     \
    __has_builtin(__builtin_amdgcn_s_wait_asynccnt)
#define HAVE_ASYNC 1
#else
#define HAVE_ASYNC 0
#endif

#if HAVE_ASYNC
// builtin signature (from hipcc diagnostic): (v4i AS1*, v4i AS3*, Imm int, Imm int)
typedef int v4i_raw __attribute__((vector_size(16)));
typedef __attribute__((address_space(1))) v4i_raw* as1_v4i_p;
typedef __attribute__((address_space(3))) v4i_raw* as3_v4i_p;
#define CP16(lds, gl)                                                    \
  __builtin_amdgcn_global_load_async_to_lds_b128(                        \
      (as1_v4i_p)(uintptr_t)(gl),                                        \
      (as3_v4i_p)(unsigned int)(uintptr_t)(lds), 0, 0)
#define ASYNC_WAIT() __builtin_amdgcn_s_wait_asynccnt(0)
#else
#define CP16(lds, gl) (*(u32x4*)(lds) = *(const u32x4*)(gl))
#define ASYNC_WAIT() ((void)0)
#endif

// ---------- helpers ----------
static __device__ __forceinline__ unsigned short f2bf(float f) {
  union { float f; unsigned int u; } c; c.f = f;
  unsigned int u = c.u;
  unsigned int r = u + 0x7FFFu + ((u >> 16) & 1u);   // round-to-nearest-even
  return (unsigned short)(r >> 16);
}

static __device__ __forceinline__ v8f vzero8() {
  v8f z;
#pragma unroll
  for (int i = 0; i < 8; ++i) z[i] = 0.0f;
  return z;
}

static __device__ __forceinline__ v8f wmma_bf16(v16bf a, v16bf b, v8f c) {
  return __builtin_amdgcn_wmma_f32_16x16x32_bf16(false, a, false, b, (short)0, c,
                                                 false, false);
}

union AF { v16bf v; u32x4 q[2]; };

// ---------- 1) group-norm statistics: one block per (b, g) ----------
__global__ void __launch_bounds__(256)
gn_stats_kernel(const float* __restrict__ x, float* __restrict__ stats) {
  __shared__ float s1[256], s2[256];
  int bg = blockIdx.x;              // b*32 + g
  int b = bg >> 5, g = bg & 31;
  const float* base = x + (size_t)b * HW_ * C_ + g * 16;
  float s = 0.f, ss = 0.f;
  for (int it = 0; it < 256; ++it) {
    int flat = it * 256 + threadIdx.x;     // 0..65535
    int p = flat >> 4, i = flat & 15;
    float v = base[(size_t)p * C_ + i];
    s += v; ss += v * v;
  }
  s1[threadIdx.x] = s; s2[threadIdx.x] = ss;
  __syncthreads();
  for (int off = 128; off > 0; off >>= 1) {
    if ((int)threadIdx.x < off) {
      s1[threadIdx.x] += s1[threadIdx.x + off];
      s2[threadIdx.x] += s2[threadIdx.x + off];
    }
    __syncthreads();
  }
  if (threadIdx.x == 0) {
    float mean = s1[0] * (1.0f / 65536.0f);
    float var  = s2[0] * (1.0f / 65536.0f) - mean * mean;
    stats[bg * 2 + 0] = mean;
    stats[bg * 2 + 1] = rsqrtf(var + 1e-5f);
  }
}

// ---------- 2) normalize + affine -> bf16 h ----------
__global__ void __launch_bounds__(256)
gn_apply_kernel(const float* __restrict__ x, const float* __restrict__ stats,
                const float* __restrict__ gsc, const float* __restrict__ gbi,
                ushort_t* __restrict__ h) {
  int idx4 = blockIdx.x * 256 + threadIdx.x;
  size_t e = (size_t)idx4 * 4;
  int m  = (int)(e >> 9);
  int c0 = (int)(e & 511);
  int b  = m >> 12;
  int g  = c0 >> 4;
  const float* st = stats + (size_t)(b * GROUPS_ + g) * 2;
  float mean = st[0], rstd = st[1];
  unsigned short us[4];
#pragma unroll
  for (int i = 0; i < 4; ++i) {
    float xv = x[e + i];
    float nv = (xv - mean) * rstd * gsc[c0 + i] + gbi[c0 + i];
    us[i] = f2bf(nv);
  }
  u32x2 pv;
  pv.x = (unsigned int)us[0] | ((unsigned int)us[1] << 16);
  pv.y = (unsigned int)us[2] | ((unsigned int)us[3] << 16);
  *(u32x2*)&h[e] = pv;
}

// ---------- 3) weights -> bf16 transposed: Wt[n][k] = W[k][n] ----------
__global__ void __launch_bounds__(256)
wconv_kernel(const float* __restrict__ wq, const float* __restrict__ wk,
             const float* __restrict__ wv, const float* __restrict__ wo,
             ushort_t* __restrict__ dst) {
  const float* src = (blockIdx.y == 0) ? wq : (blockIdx.y == 1) ? wk
                   : (blockIdx.y == 2) ? wv : wo;
  ushort_t* d = dst + (size_t)blockIdx.y * C_ * C_;
  int idx = blockIdx.x * 256 + threadIdx.x;   // idx = n*512 + k
  int n = idx >> 9, kk = idx & 511;
  d[idx] = f2bf(src[(size_t)kk * C_ + n]);
}

// ---------- 4) GEMM (double-buffered LDS, async staging) ----------
// grid (M/64, N/64); 128 threads = 4 waves; wave -> 32x32 (2x2 WMMA tiles)
// vtrans=1: store bf16 output transposed as outT[n][m] (for V)
__global__ void __launch_bounds__(128)
gemm_kernel(const ushort_t* __restrict__ A, const ushort_t* __restrict__ Wt,
            const float* __restrict__ bias, float alpha, int vtrans,
            ushort_t* __restrict__ outb, float* __restrict__ outf,
            const float* __restrict__ resid) {
  __shared__ ushort_t As[2][64 * 40];   // padded rows: 40 bf16 = 20 dwords
  __shared__ ushort_t Bs[2][64 * 40];
  int tid = threadIdx.x;
  int w = tid >> 5, lane = tid & 31, hl = lane >> 4, ln = lane & 15;
  int m0 = blockIdx.x * 64, n0 = blockIdx.y * 64;
  int wm = (w >> 1) * 32, wn = (w & 1) * 32;
  int kba = hl ? 8 : 0;
  int kbb = hl ? 8 : 0;
  int sr = tid >> 2, sco = (tid & 3) * 8;            // staging coords
  const ushort_t* Ag = A  + (size_t)(m0 + sr) * C_ + sco;
  const ushort_t* Wg = Wt + (size_t)(n0 + sr) * C_ + sco;

#define GSTAGE(buf, kt)                                                        \
  do {                                                                         \
    CP16(&As[buf][sr * 40 + sco],        Ag + (kt));                           \
    CP16(&As[buf][(sr + 32) * 40 + sco], Ag + (size_t)32 * C_ + (kt));         \
    CP16(&Bs[buf][sr * 40 + sco],        Wg + (kt));                           \
    CP16(&Bs[buf][(sr + 32) * 40 + sco], Wg + (size_t)32 * C_ + (kt));         \
  } while (0)

  v8f acc[2][2];
  acc[0][0] = vzero8(); acc[0][1] = vzero8();
  acc[1][0] = vzero8(); acc[1][1] = vzero8();

  GSTAGE(0, 0);
  ASYNC_WAIT();
  __syncthreads();

  for (int t = 0; t < 16; ++t) {
    int cur = t & 1;
    if (t + 1 < 16) GSTAGE(cur ^ 1, (t + 1) * 32);
    AF a[2], bb[2];
#pragma unroll
    for (int i = 0; i < 2; ++i) {
      a[i].q[0]  = *(const u32x4*)&As[cur][(wm + i * 16 + ln) * 40 + kba];
      a[i].q[1]  = *(const u32x4*)&As[cur][(wm + i * 16 + ln) * 40 + kba + 16];
      bb[i].q[0] = *(const u32x4*)&Bs[cur][(wn + i * 16 + ln) * 40 + kbb];
      bb[i].q[1] = *(const u32x4*)&Bs[cur][(wn + i * 16 + ln) * 40 + kbb + 16];
    }
#pragma unroll
    for (int i = 0; i < 2; ++i)
#pragma unroll
      for (int j = 0; j < 2; ++j)
        acc[i][j] = wmma_bf16(a[i].v, bb[j].v, acc[i][j]);
    ASYNC_WAIT();
    __syncthreads();
  }
#undef GSTAGE

#pragma unroll
  for (int i = 0; i < 2; ++i) {
    int row0 = m0 + wm + i * 16 + 8 * hl;
#pragma unroll
    for (int j = 0; j < 2; ++j) {
      int col = n0 + wn + j * 16 + ln;
      float bv = bias[col];
      if (outf) {
#pragma unroll
        for (int r = 0; r < 8; ++r) {
          size_t idx = (size_t)(row0 + r) * C_ + col;
          outf[idx] = acc[i][j][r] + bv + resid[idx];
        }
      } else if (vtrans) {
        unsigned short us[8];
#pragma unroll
        for (int r = 0; r < 8; ++r) us[r] = f2bf((acc[i][j][r] + bv) * alpha);
        u32x4 pk;
        pk.x = (unsigned int)us[0] | ((unsigned int)us[1] << 16);
        pk.y = (unsigned int)us[2] | ((unsigned int)us[3] << 16);
        pk.z = (unsigned int)us[4] | ((unsigned int)us[5] << 16);
        pk.w = (unsigned int)us[6] | ((unsigned int)us[7] << 16);
        *(u32x4*)&outb[(size_t)col * MTOT + row0] = pk;     // outT[n][m]
      } else {
#pragma unroll
        for (int r = 0; r < 8; ++r)
          outb[(size_t)(row0 + r) * C_ + col] = f2bf((acc[i][j][r] + bv) * alpha);
      }
    }
  }
}

// ---------- 5) flash attention: grid (HW/64, B); 128 threads = 4 waves ----------
// Kc = 64 keys/iter.  O accumulated TRANSPOSED in LDS: Ot[d][query] (stride 68)
// so the WMMA C operand is 8 contiguous floats per lane (2 x ds_load_b128).
#define NKEY  64
#define KSTR  520
#define VSTR  72
#define PSTR  72
#define OTSTR 68
__global__ void __launch_bounds__(128)
attn_kernel(const ushort_t* __restrict__ Q, const ushort_t* __restrict__ K,
            const ushort_t* __restrict__ VT, ushort_t* __restrict__ Oout) {
  extern __shared__ char smem[];
  float*    Ot = (float*)smem;                                    // [512][68]
  ushort_t* Ks = (ushort_t*)(smem + 512 * OTSTR * 4);             // [64][520]
  ushort_t* Vs = Ks + 64 * KSTR;                                  // [512][72]
  ushort_t* Ps = Vs + 512 * VSTR;                                 // [4][16][72]
  float*    Lr = (float*)(Ps + 4 * 16 * PSTR);                    // [64]

  int b = blockIdx.y;
  int q0 = blockIdx.x * 64;
  int tid = threadIdx.x;
  int w = tid >> 5, lane = tid & 31, hl = lane >> 4, ln = lane & 15;
  int kba = hl ? 8 : 0;
  int kbb = hl ? 8 : 0;

  const ushort_t* Qb = Q  + (size_t)b * HW_ * C_;
  const ushort_t* Kb = K  + (size_t)b * HW_ * C_;
  const ushort_t* Vb = VT + (size_t)b * HW_;      // vT[d][m] : d*MTOT + b*HW_ + key

  for (int i = tid; i < 512 * OTSTR; i += 128) Ot[i] = 0.0f;

  // Q fragments live in registers: rows q0 + w*16 + ln, 16 chunks of K=32
  AF qf[16];
  const ushort_t* qp = Qb + (size_t)(q0 + w * 16 + ln) * C_;
#pragma unroll
  for (int c = 0; c < 16; ++c) {
    qf[c].q[0] = *(const u32x4*)&qp[c * 32 + kba];
    qf[c].q[1] = *(const u32x4*)&qp[c * 32 + kba + 16];
  }

  float mrun[8], lrun[8];
#pragma unroll
  for (int r = 0; r < 8; ++r) { mrun[r] = -1e30f; lrun[r] = 0.0f; }

  ushort_t* Pw = Ps + w * 16 * PSTR;
  int orow0 = w * 16 + 8 * hl;

  for (int j = 0; j < HW_ / NKEY; ++j) {
    int k0 = j * NKEY;
    // stage K row-major + V (already transposed in global) ; prefetch next tile
#pragma unroll 4
    for (int it = 0; it < 32; ++it) {
      int ch = it * 128 + tid;                  // 0..4095
      int krow = ch >> 6, kco = (ch & 63) * 8;
      const ushort_t* ksrc = Kb + (size_t)(k0 + krow) * C_ + kco;
      CP16(&Ks[krow * KSTR + kco], ksrc);
      int vd = ch >> 3, vco = (ch & 7) * 8;
      const ushort_t* vsrc = Vb + (size_t)vd * MTOT + k0 + vco;
      CP16(&Vs[vd * VSTR + vco], vsrc);
      if (j + 1 < HW_ / NKEY) {
        __builtin_prefetch(ksrc + (size_t)NKEY * C_);
        __builtin_prefetch(vsrc + NKEY);
      }
    }
    ASYNC_WAIT();
    __syncthreads();

    // S = Q * K^T over 4 key-tiles of 16
    v8f s[4];
    s[0] = vzero8(); s[1] = vzero8(); s[2] = vzero8(); s[3] = vzero8();
#pragma unroll
    for (int c = 0; c < 16; ++c) {
#pragma unroll
      for (int t = 0; t < 4; ++t) {
        AF bt;
        bt.q[0] = *(const u32x4*)&Ks[(t * 16 + ln) * KSTR + c * 32 + kbb];
        bt.q[1] = *(const u32x4*)&Ks[(t * 16 + ln) * KSTR + c * 32 + kbb + 16];
        s[t] = wmma_bf16(qf[c].v, bt.v, s[t]);
      }
    }

    // online softmax (rows r + 8*hl, columns across 16 lanes of the half)
    float alr[8];
#pragma unroll
    for (int r = 0; r < 8; ++r) {
      float e0 = s[0][r], e1 = s[1][r], e2 = s[2][r], e3 = s[3][r];
      float mx = fmaxf(fmaxf(e0, e1), fmaxf(e2, e3));
#pragma unroll
      for (int off = 8; off >= 1; off >>= 1) mx = fmaxf(mx, __shfl_xor(mx, off, 32));
      float mn = fmaxf(mrun[r], mx);
      float p0 = __expf(e0 - mn), p1 = __expf(e1 - mn);
      float p2 = __expf(e2 - mn), p3 = __expf(e3 - mn);
      float sm = (p0 + p1) + (p2 + p3);
#pragma unroll
      for (int off = 8; off >= 1; off >>= 1) sm += __shfl_xor(sm, off, 32);
      float al = __expf(mrun[r] - mn);
      lrun[r] = lrun[r] * al + sm;
      mrun[r] = mn;
      alr[r] = al;
      int prow = r + 8 * hl;
      Pw[prow * PSTR + 0  + ln] = f2bf(p0);
      Pw[prow * PSTR + 16 + ln] = f2bf(p1);
      Pw[prow * PSTR + 32 + ln] = f2bf(p2);
      Pw[prow * PSTR + 48 + ln] = f2bf(p3);
    }

    // P as two A-fragments (16 queries x 64 keys)
    AF pf0, pf1;
    pf0.q[0] = *(const u32x4*)&Pw[ln * PSTR + kba];
    pf0.q[1] = *(const u32x4*)&Pw[ln * PSTR + kba + 16];
    pf1.q[0] = *(const u32x4*)&Pw[ln * PSTR + 32 + kba];
    pf1.q[1] = *(const u32x4*)&Pw[ln * PSTR + 32 + kba + 16];

    // O += P * V over 32 d-tiles; rescale folded into the C operand
#pragma unroll 4
    for (int nt = 0; nt < 32; ++nt) {
      int d = nt * 16 + ln;
      AF bv0, bv1;
      bv0.q[0] = *(const u32x4*)&Vs[d * VSTR + kbb];
      bv0.q[1] = *(const u32x4*)&Vs[d * VSTR + kbb + 16];
      bv1.q[0] = *(const u32x4*)&Vs[d * VSTR + 32 + kbb];
      bv1.q[1] = *(const u32x4*)&Vs[d * VSTR + 32 + kbb + 16];
      union { v8f f; u32x4 q[2]; } cc;
      cc.q[0] = *(const u32x4*)&Ot[d * OTSTR + orow0];
      cc.q[1] = *(const u32x4*)&Ot[d * OTSTR + orow0 + 4];
#pragma unroll
      for (int r = 0; r < 8; ++r) cc.f[r] *= alr[r];
      cc.f = wmma_bf16(pf0.v, bv0.v, cc.f);
      cc.f = wmma_bf16(pf1.v, bv1.v, cc.f);
      *(u32x4*)&Ot[d * OTSTR + orow0]     = cc.q[0];
      *(u32x4*)&Ot[d * OTSTR + orow0 + 4] = cc.q[1];
    }
    __syncthreads();
  }

  if (ln == 0) {
#pragma unroll
    for (int r = 0; r < 8; ++r) Lr[w * 16 + 8 * hl + r] = lrun[r];
  }
  __syncthreads();

  for (int i = tid; i < 64 * C_; i += 128) {
    int row = i >> 9, col = i & 511;
    float val = Ot[col * OTSTR + row] / Lr[row];
    Oout[((size_t)b * HW_ + q0 + row) * C_ + col] = f2bf(val);
  }
}

// ---------- host launch ----------
extern "C" void kernel_launch(void* const* d_in, const int* in_sizes, int n_in,
                              void* d_out, int out_size, void* d_ws, size_t ws_size,
                              hipStream_t stream) {
  (void)in_sizes; (void)n_in; (void)out_size; (void)ws_size;
  const float* x   = (const float*)d_in[0];
  const float* gsc = (const float*)d_in[1];
  const float* gbi = (const float*)d_in[2];
  const float* wq  = (const float*)d_in[3];
  const float* bq  = (const float*)d_in[4];
  const float* wk  = (const float*)d_in[5];
  const float* bk  = (const float*)d_in[6];
  const float* wv  = (const float*)d_in[7];
  const float* bv  = (const float*)d_in[8];
  const float* wo  = (const float*)d_in[9];
  const float* bo  = (const float*)d_in[10];
  float* out = (float*)d_out;

  char* ws = (char*)d_ws;
  const size_t SZ_H = (size_t)MTOT * C_ * 2;             // 32 MB bf16 buffer
  float*    stats = (float*)ws;                           // 2 KB
  ushort_t* h     = (ushort_t*)(ws + 4096);
  ushort_t* qbuf  = (ushort_t*)(ws + 4096 + 1 * SZ_H);
  ushort_t* kbuf  = (ushort_t*)(ws + 4096 + 2 * SZ_H);
  ushort_t* vbuf  = (ushort_t*)(ws + 4096 + 3 * SZ_H);   // holds vT[n][m]
  ushort_t* wtb   = (ushort_t*)(ws + 4096 + 4 * SZ_H);   // 4 x 512 KB bf16 W^T
  ushort_t* attnb = h;                                    // h dead after QKV

  ushort_t* wqT = wtb;
  ushort_t* wkT = wtb + 1 * C_ * C_;
  ushort_t* wvT = wtb + 2 * C_ * C_;
  ushort_t* woT = wtb + 3 * C_ * C_;

  gn_stats_kernel<<<dim3(BATCH_ * GROUPS_), 256, 0, stream>>>(x, stats);
  gn_apply_kernel<<<dim3((MTOT * C_ / 4) / 256), 256, 0, stream>>>(
      x, stats, gsc, gbi, h);
  wconv_kernel<<<dim3(C_ * C_ / 256, 4), 256, 0, stream>>>(wq, wk, wv, wo, wtb);

  const float qscale = 0.044194173824159216f;  // 512^-0.5 folded into Q
  dim3 ggrid(MTOT / 64, C_ / 64);
  gemm_kernel<<<ggrid, 128, 0, stream>>>(h, wqT, bq, qscale, 0, qbuf, nullptr, nullptr);
  gemm_kernel<<<ggrid, 128, 0, stream>>>(h, wkT, bk, 1.0f, 0, kbuf, nullptr, nullptr);
  gemm_kernel<<<ggrid, 128, 0, stream>>>(h, wvT, bv, 1.0f, 1, vbuf, nullptr, nullptr);

  const size_t attn_smem = (size_t)512 * OTSTR * 4 + 64 * KSTR * 2 +
                           512 * VSTR * 2 + 4 * 16 * PSTR * 2 + 64 * 4;  // ~282 KB
  attn_kernel<<<dim3(HW_ / 64, BATCH_), 128, attn_smem, stream>>>(qbuf, kbuf, vbuf, attnb);

  gemm_kernel<<<ggrid, 128, 0, stream>>>(attnb, woT, bo, 1.0f, 0, nullptr, out, x);
}
